// DynamicSparseAttention_33827162423567
// MI455X (gfx1250) — matrix operand order repaired and checked by reading, not compiled, not run on hardware
//
#include <hip/hip_runtime.h>
#include <hip/hip_bf16.h>
#include <math.h>

typedef __attribute__((ext_vector_type(16))) _Float16 v16h;
typedef __attribute__((ext_vector_type(8)))  _Float16 v8h;
typedef __attribute__((ext_vector_type(8)))  float    v8f;

#define NB     2
#define DIMC   256
#define NHEADS 8
#define HDIM   32
#define NPIX   1600      // 40*40
#define HH     40
#define WW     40
#define NBH    (NB*NHEADS)
#define SCALE  0.17677669529663687f   // 1/sqrt(32)

union V16U { v16h v; v8h h[2]; };

// A-matrix 16x32 f16 per-lane fragment: lane<16 holds K=0..15 of row M=lane;
// lane>=16 holds K=16..31 of row M=lane-16.  Two 16-byte loads.
__device__ __forceinline__ v16h load_afrag(const _Float16* row, int kOffA) {
  V16U u;
  u.h[0] = *(const v8h*)(row + 2 * kOffA);
  u.h[1] = *(const v8h*)(row + 2 * kOffA + 8);
  return u.v;
}
// B-matrix 32x16 f16: lane<16 holds K=0..15 of col N=lane; lane>=16 K=16..31.
__device__ __forceinline__ v16h load_bfrag(const _Float16* p) {
  V16U u;
  u.h[0] = *(const v8h*)(p);
  u.h[1] = *(const v8h*)(p + 8);
  return u.v;
}
#define WMMA(a, b, c) __builtin_amdgcn_wmma_f32_16x16x32_f16(false, (a), false, (b), (short)0, (c), false, false)

// ---------------- elementwise / pack kernels ----------------

__global__ void k_cvt_f16(const float* __restrict__ src, _Float16* __restrict__ dst, int n) {
  int i = blockIdx.x * blockDim.x + threadIdx.x;
  if (i < n) dst[i] = (_Float16)src[i];
}

// x [b][ic][p] f32 -> xt [b][p][ic] f16
__global__ void k_pack_xt(const float* __restrict__ x, _Float16* __restrict__ xt) {
  int i = blockIdx.x * blockDim.x + threadIdx.x;
  if (i >= NB * DIMC * NPIX) return;
  int p  = i % NPIX;
  int t  = i / NPIX;
  int ic = t % DIMC;
  int b  = t / DIMC;
  xt[((size_t)b * NPIX + p) * DIMC + ic] = (_Float16)x[((size_t)b * DIMC + ic) * NPIX + p];
}

// depthwise 3x3, pad=1: qkv_dw[b][ch][p]
__global__ void k_dwconv(const float* __restrict__ src, const float* __restrict__ pw,
                         const float* __restrict__ pb, float* __restrict__ dst) {
  int i = blockIdx.x * blockDim.x + threadIdx.x;
  if (i >= NB * 3 * DIMC * NPIX) return;
  int p  = i % NPIX;
  int t  = i / NPIX;
  int ch = t % (3 * DIMC);
  int b  = t / (3 * DIMC);
  int y = p / WW, x = p % WW;
  const float* w = pw + ch * 9;
  float s = pb[ch];
  const float* base = src + ((size_t)b * 3 * DIMC + ch) * NPIX;
#pragma unroll
  for (int dy = -1; dy <= 1; ++dy) {
#pragma unroll
    for (int dx = -1; dx <= 1; ++dx) {
      int yy = y + dy, xx = x + dx;
      if (yy >= 0 && yy < HH && xx >= 0 && xx < WW)
        s += base[yy * WW + xx] * w[(dy + 1) * 3 + (dx + 1)];
    }
  }
  dst[i] = s;
}

// split qkv_dw into head-major f16 operands
__global__ void k_pack_heads(const float* __restrict__ qkv, _Float16* __restrict__ q_h,
                             _Float16* __restrict__ k_h, _Float16* __restrict__ v_t) {
  int i = blockIdx.x * blockDim.x + threadIdx.x;
  if (i >= NB * DIMC * NPIX) return;
  int p = i % NPIX;
  int t = i / NPIX;
  int c = t % DIMC;
  int b = t / DIMC;
  int head = c / HDIM, d = c % HDIM;
  int bh = b * NHEADS + head;
  q_h[((size_t)bh * NPIX + p) * HDIM + d] = (_Float16)qkv[((size_t)b * 3 * DIMC + c) * NPIX + p];
  k_h[((size_t)bh * NPIX + p) * HDIM + d] = (_Float16)qkv[((size_t)b * 3 * DIMC + DIMC + c) * NPIX + p];
  v_t[((size_t)bh * HDIM + d) * NPIX + p] = (_Float16)qkv[((size_t)b * 3 * DIMC + 2 * DIMC + c) * NPIX + p];
}

// ---------------- WMMA GEMM kernels (register-blocked, wave per tile-group) --------

// qkv: wave computes 16x64 (1 M-tile x 4 N-tiles, A fragment reused 4x). K=256.
__global__ void k_gemm_qkv(const _Float16* __restrict__ wq, const _Float16* __restrict__ xt,
                           const float* __restrict__ bias, float* __restrict__ out) {
  int wid = blockIdx.x * 4 + (threadIdx.x >> 5);
  int lane = threadIdx.x & 31;
  int ng = wid % 25; int t2 = wid / 25;
  int mt = t2 % 48;  int b  = t2 / 48;
  int Mb = mt * 16, Nb = ng * 64, lm = lane & 15;
  int kOffA = (lane >= 16) ? 8 : 0;
  int kOffB = (lane >= 16) ? 16 : 0;
  const _Float16* arow = wq + (size_t)(Mb + lm) * DIMC;
  const _Float16* brow = xt + ((size_t)b * NPIX + Nb + lm) * DIMC;
  v8f c0 = {}, c1 = {}, c2 = {}, c3 = {};
#pragma unroll
  for (int k = 0; k < DIMC; k += 32) {
    v16h a  = load_afrag(arow + k, kOffA);
    v16h b0 = load_bfrag(brow + k + kOffB);
    v16h b1 = load_bfrag(brow + (size_t)16 * DIMC + k + kOffB);
    v16h b2 = load_bfrag(brow + (size_t)32 * DIMC + k + kOffB);
    v16h b3 = load_bfrag(brow + (size_t)48 * DIMC + k + kOffB);
    c0 = WMMA(a, b0, c0); c1 = WMMA(a, b1, c1);
    c2 = WMMA(a, b2, c2); c3 = WMMA(a, b3, c3);
  }
  int Mr = (lane >= 16) ? 8 : 0;
  v8f acc[4] = {c0, c1, c2, c3};
#pragma unroll
  for (int j = 0; j < 4; ++j) {
#pragma unroll
    for (int r = 0; r < 8; ++r) {
      int M = Mb + Mr + r, N = Nb + j * 16 + lm;
      out[((size_t)b * 3 * DIMC + M) * NPIX + N] = acc[j][r] + bias[M];
    }
  }
}

// scores: wave computes 16x64 strip of attn (K=32 -> 4 WMMAs, A reused)
__global__ void k_scores(const _Float16* __restrict__ q_h, const _Float16* __restrict__ k_h,
                         float* __restrict__ attn) {
  int wid = blockIdx.x * 4 + (threadIdx.x >> 5);
  int lane = threadIdx.x & 31;
  int ng = wid % 25; int t2 = wid / 25;
  int mt = t2 % 100; int bh = t2 / 100;
  int Mb = mt * 16, Nb = ng * 64, lm = lane & 15;
  int kOffA = (lane >= 16) ? 8 : 0;
  int kOffB = (lane >= 16) ? 16 : 0;
  const _Float16* arow = q_h + ((size_t)bh * NPIX + Mb + lm) * HDIM;
  const _Float16* brow = k_h + ((size_t)bh * NPIX + Nb + lm) * HDIM;
  v16h a  = load_afrag(arow, kOffA);
  v16h b0 = load_bfrag(brow + kOffB);
  v16h b1 = load_bfrag(brow + 16 * HDIM + kOffB);
  v16h b2 = load_bfrag(brow + 32 * HDIM + kOffB);
  v16h b3 = load_bfrag(brow + 48 * HDIM + kOffB);
  v8f c0 = {}, c1 = {}, c2 = {}, c3 = {};
  c0 = WMMA(a, b0, c0); c1 = WMMA(a, b1, c1);
  c2 = WMMA(a, b2, c2); c3 = WMMA(a, b3, c3);
  int Mr = (lane >= 16) ? 8 : 0;
  v8f acc[4] = {c0, c1, c2, c3};
#pragma unroll
  for (int j = 0; j < 4; ++j) {
#pragma unroll
    for (int r = 0; r < 8; ++r) {
      int M = Mb + Mr + r, N = Nb + j * 16 + lm;
      attn[((size_t)bh * NPIX + M) * NPIX + N] = acc[j][r] * SCALE;
    }
  }
}

// ctx = P @ V : wave computes 32x32 (2x2 tiles), K=1600
__global__ void k_pv(const _Float16* __restrict__ p_h, const _Float16* __restrict__ v_t,
                     _Float16* __restrict__ ctx_h) {
  int wid = blockIdx.x * 4 + (threadIdx.x >> 5);
  int lane = threadIdx.x & 31;
  int mg = wid % 50; int bh = wid / 50;
  int Mb = mg * 32, lm = lane & 15;
  int kOffA = (lane >= 16) ? 8 : 0;
  int kOffB = (lane >= 16) ? 16 : 0;
  const _Float16* arow0 = p_h + ((size_t)bh * NPIX + Mb + lm) * NPIX;
  const _Float16* arow1 = arow0 + (size_t)16 * NPIX;
  const _Float16* brow0 = v_t + ((size_t)bh * HDIM + lm) * NPIX;
  const _Float16* brow1 = brow0 + (size_t)16 * NPIX;
  v8f c00 = {}, c01 = {}, c10 = {}, c11 = {};
  for (int k = 0; k < NPIX; k += 32) {
    v16h a0 = load_afrag(arow0 + k, kOffA);
    v16h a1 = load_afrag(arow1 + k, kOffA);
    v16h b0 = load_bfrag(brow0 + k + kOffB);
    v16h b1 = load_bfrag(brow1 + k + kOffB);
    c00 = WMMA(a0, b0, c00); c01 = WMMA(a0, b1, c01);
    c10 = WMMA(a1, b0, c10); c11 = WMMA(a1, b1, c11);
  }
  int b = bh / NHEADS, head = bh % NHEADS;
  int Mr = (lane >= 16) ? 8 : 0;
  v8f acc[2][2] = {{c00, c01}, {c10, c11}};
#pragma unroll
  for (int mi = 0; mi < 2; ++mi) {
#pragma unroll
    for (int nj = 0; nj < 2; ++nj) {
#pragma unroll
      for (int r = 0; r < 8; ++r) {
        int M = Mb + mi * 16 + Mr + r, N = nj * 16 + lm;
        ctx_h[((size_t)b * NPIX + M) * DIMC + head * HDIM + N] = (_Float16)acc[mi][nj][r];
      }
    }
  }
}

// proj: wave computes 16x64, K=256
__global__ void k_proj(const _Float16* __restrict__ wp, const _Float16* __restrict__ ctx_h,
                       const float* __restrict__ pb, float* __restrict__ out) {
  int wid = blockIdx.x * 4 + (threadIdx.x >> 5);
  int lane = threadIdx.x & 31;
  int ng = wid % 25; int t2 = wid / 25;
  int mt = t2 % 16;  int b  = t2 / 16;
  int Mb = mt * 16, Nb = ng * 64, lm = lane & 15;
  int kOffA = (lane >= 16) ? 8 : 0;
  int kOffB = (lane >= 16) ? 16 : 0;
  const _Float16* arow = wp + (size_t)(Mb + lm) * DIMC;
  const _Float16* brow = ctx_h + ((size_t)b * NPIX + Nb + lm) * DIMC;
  v8f c0 = {}, c1 = {}, c2 = {}, c3 = {};
#pragma unroll
  for (int k = 0; k < DIMC; k += 32) {
    v16h a  = load_afrag(arow + k, kOffA);
    v16h b0 = load_bfrag(brow + k + kOffB);
    v16h b1 = load_bfrag(brow + (size_t)16 * DIMC + k + kOffB);
    v16h b2 = load_bfrag(brow + (size_t)32 * DIMC + k + kOffB);
    v16h b3 = load_bfrag(brow + (size_t)48 * DIMC + k + kOffB);
    c0 = WMMA(a, b0, c0); c1 = WMMA(a, b1, c1);
    c2 = WMMA(a, b2, c2); c3 = WMMA(a, b3, c3);
  }
  int Mr = (lane >= 16) ? 8 : 0;
  v8f acc[4] = {c0, c1, c2, c3};
#pragma unroll
  for (int j = 0; j < 4; ++j) {
#pragma unroll
    for (int r = 0; r < 8; ++r) {
      int M = Mb + Mr + r, N = Nb + j * 16 + lm;
      out[((size_t)b * DIMC + M) * NPIX + N] = acc[j][r] + pb[M];
    }
  }
}

// ---------------- entropy / gate / fused top-k + masked softmax ----------------

// per-row entropy of softmax(attn_row): H = log Z - S/Z
__global__ void k_entropy(const float* __restrict__ attn, float* __restrict__ ent) {
  int row = blockIdx.x;
  const float* s = attn + (size_t)row * NPIX;
  __shared__ float r1[256], r2[256];
  int t = threadIdx.x;
  float m = -INFINITY;
  for (int i = t; i < NPIX; i += 256) m = fmaxf(m, s[i]);
  r1[t] = m; __syncthreads();
  for (int st = 128; st > 0; st >>= 1) { if (t < st) r1[t] = fmaxf(r1[t], r1[t + st]); __syncthreads(); }
  m = r1[0]; __syncthreads();
  float z = 0.f, sw = 0.f;
  for (int i = t; i < NPIX; i += 256) { float d = s[i] - m; float e = __expf(d); z += e; sw += e * d; }
  r1[t] = z; r2[t] = sw; __syncthreads();
  for (int st = 128; st > 0; st >>= 1) { if (t < st) { r1[t] += r1[t + st]; r2[t] += r2[t + st]; } __syncthreads(); }
  if (t == 0) ent[row] = logf(r1[0]) - r2[0] / r1[0];
}

__global__ void k_gate(const float* __restrict__ ent, const float* __restrict__ g1w,
                       const float* __restrict__ g1b, const float* __restrict__ g2w,
                       const float* __restrict__ g2b, int* __restrict__ keep) {
  int bh = blockIdx.x, t = threadIdx.x;
  __shared__ float r1[256];
  float s = 0.f;
  for (int i = t; i < NPIX; i += 256) s += ent[(size_t)bh * NPIX + i];
  r1[t] = s; __syncthreads();
  for (int st = 128; st > 0; st >>= 1) { if (t < st) r1[t] += r1[t + st]; __syncthreads(); }
  if (t == 0) {
    float e = r1[0] / (float)NPIX;
    float acc = g2b[0];
#pragma unroll
    for (int j = 0; j < 16; ++j) {
      float h = fmaxf(e * g1w[j] + g1b[j], 0.f);
      acc += g2w[j] * h;
    }
    float ratio = 0.9f / (1.f + __expf(-acc)) + 0.1f;
    int kp = (int)ceilf(ratio * (float)NPIX);
    keep[bh] = min(NPIX, max(1, kp));
  }
}

// Fused: async-DMA the score row into LDS once, exact keep-th-largest threshold by
// bisection on order-preserving uint keys, then masked softmax -> f16 probs.
__global__ void k_select_softmax(const float* __restrict__ attn, const int* __restrict__ keep,
                                 _Float16* __restrict__ p_h) {
  int row = blockIdx.x, t = threadIdx.x;
  __shared__ float    srow[NPIX];
  __shared__ unsigned keys[NPIX];
  __shared__ float    r1[256];
  __shared__ int cnt;
  __shared__ unsigned s_lo, s_hi;

  const float* s = attn + (size_t)row * NPIX;

  // stage row (6400 B = 400 x b128) into LDS via CDNA5 async-to-LDS DMA
  unsigned ldsbase = (unsigned)(unsigned long long)(uintptr_t)&srow[0];
  for (int idx = t; idx < NPIX / 4; idx += 256) {
    unsigned lds = ldsbase + (unsigned)idx * 16u;
    unsigned long long ga = (unsigned long long)(uintptr_t)(s + idx * 4);
    asm volatile("global_load_async_to_lds_b128 %0, %1, off" :: "v"(lds), "v"(ga) : "memory");
  }
  asm volatile("s_wait_asynccnt 0" ::: "memory");
  __syncthreads();

  // order-preserving uint keys
  for (int i = t; i < NPIX; i += 256) {
    int bits = __float_as_int(srow[i]);
    keys[i] = (bits < 0) ? ~(unsigned)bits : ((unsigned)bits | 0x80000000u);
  }
  int kp = keep[row / NPIX];
  if (t == 0) { s_lo = 0u; s_hi = 0xFFFFFFFFu; }
  __syncthreads();
  for (int it = 0; it < 33; ++it) {
    unsigned lo = s_lo, hi = s_hi;
    if (lo >= hi) break;                       // uniform
    unsigned mid = lo + ((hi - lo) >> 1) + 1;  // candidate in (lo, hi]
    if (t == 0) cnt = 0;
    __syncthreads();
    int local = 0;
    for (int i = t; i < NPIX; i += 256) local += (keys[i] >= mid) ? 1 : 0;
    atomicAdd(&cnt, local);
    __syncthreads();
    if (t == 0) { if (cnt >= kp) s_lo = mid; else s_hi = mid - 1; }
    __syncthreads();
  }
  unsigned u = s_lo;
  int kbits = (u & 0x80000000u) ? (int)(u & 0x7FFFFFFFu) : ~(int)u;
  float th = __int_as_float(kbits);

  // masked softmax from LDS
  float m = -INFINITY;
  for (int i = t; i < NPIX; i += 256) { float v = srow[i]; if (v >= th) m = fmaxf(m, v); }
  r1[t] = m; __syncthreads();
  for (int st = 128; st > 0; st >>= 1) { if (t < st) r1[t] = fmaxf(r1[t], r1[t + st]); __syncthreads(); }
  m = r1[0]; __syncthreads();
  float z = 0.f;
  for (int i = t; i < NPIX; i += 256) { float v = srow[i]; if (v >= th) z += __expf(v - m); }
  r1[t] = z; __syncthreads();
  for (int st = 128; st > 0; st >>= 1) { if (t < st) r1[t] += r1[t + st]; __syncthreads(); }
  float inv = 1.f / r1[0];
  for (int i = t; i < NPIX; i += 256) {
    float v = srow[i];
    float pv = (v >= th) ? __expf(v - m) * inv : 0.f;
    p_h[(size_t)row * NPIX + i] = (_Float16)pv;
  }
}

// ---------------- launcher ----------------

static inline size_t alignup(size_t x) { return (x + 255) & ~(size_t)255; }

extern "C" void kernel_launch(void* const* d_in, const int* in_sizes, int n_in,
                              void* d_out, int out_size, void* d_ws, size_t ws_size,
                              hipStream_t stream) {
  const float* x      = (const float*)d_in[0];
  const float* qkv_w  = (const float*)d_in[1];
  const float* qkv_b  = (const float*)d_in[2];
  const float* pos_w  = (const float*)d_in[3];
  const float* pos_b  = (const float*)d_in[4];
  const float* proj_w = (const float*)d_in[5];
  const float* proj_b = (const float*)d_in[6];
  const float* g1_w   = (const float*)d_in[7];
  const float* g1_b   = (const float*)d_in[8];
  const float* g2_w   = (const float*)d_in[9];
  const float* g2_b   = (const float*)d_in[10];
  float* out = (float*)d_out;

  char* ws = (char*)d_ws;
  size_t off = 0;
  _Float16* xt_h    = (_Float16*)(ws + off); off = alignup(off + (size_t)NB * NPIX * DIMC * 2);
  _Float16* wq_h    = (_Float16*)(ws + off); off = alignup(off + (size_t)3 * DIMC * DIMC * 2);
  _Float16* wp_h    = (_Float16*)(ws + off); off = alignup(off + (size_t)DIMC * DIMC * 2);
  float*    qkv_pre = (float*)   (ws + off); off = alignup(off + (size_t)NB * 3 * DIMC * NPIX * 4);
  float*    qkv_dw  = (float*)   (ws + off); off = alignup(off + (size_t)NB * 3 * DIMC * NPIX * 4);
  _Float16* q_h     = (_Float16*)(ws + off); off = alignup(off + (size_t)NBH * NPIX * HDIM * 2);
  _Float16* k_hh    = (_Float16*)(ws + off); off = alignup(off + (size_t)NBH * NPIX * HDIM * 2);
  _Float16* v_t     = (_Float16*)(ws + off); off = alignup(off + (size_t)NBH * HDIM * NPIX * 2);
  float*    attn    = (float*)   (ws + off); off = alignup(off + (size_t)NBH * NPIX * NPIX * 4);
  float*    ent     = (float*)   (ws + off); off = alignup(off + (size_t)NBH * NPIX * 4);
  int*      keep    = (int*)     (ws + off); off = alignup(off + (size_t)NBH * 4);
  _Float16* p_h     = (_Float16*)(ws + off); off = alignup(off + (size_t)NBH * NPIX * NPIX * 2);
  _Float16* ctx_h   = (_Float16*)(ws + off); off = alignup(off + (size_t)NB * NPIX * DIMC * 2);
  (void)ws_size; (void)in_sizes; (void)n_in; (void)out_size;

  // 1. pack / convert operands to f16
  k_pack_xt<<<(NB * DIMC * NPIX + 255) / 256, 256, 0, stream>>>(x, xt_h);
  k_cvt_f16<<<(3 * DIMC * DIMC + 255) / 256, 256, 0, stream>>>(qkv_w, wq_h, 3 * DIMC * DIMC);
  k_cvt_f16<<<(DIMC * DIMC + 255) / 256, 256, 0, stream>>>(proj_w, wp_h, DIMC * DIMC);

  // 2. qkv 1x1 conv as WMMA GEMM: 2*48*25 tile-groups, 4 waves/block
  k_gemm_qkv<<<(NB * 48 * 25) / 4, 128, 0, stream>>>(wq_h, xt_h, qkv_b, qkv_pre);

  // 3. depthwise 3x3
  k_dwconv<<<(NB * 3 * DIMC * NPIX + 255) / 256, 256, 0, stream>>>(qkv_pre, pos_w, pos_b, qkv_dw);

  // 4. split into head-major f16 operands
  k_pack_heads<<<(NB * DIMC * NPIX + 255) / 256, 256, 0, stream>>>(qkv_dw, q_h, k_hh, v_t);

  // 5. attention scores: 16 * 100 * 25 tile-groups
  k_scores<<<(NBH * 100 * 25) / 4, 128, 0, stream>>>(q_h, k_hh, attn);

  // 6. entropy per row, gate per (b,h)
  k_entropy<<<NBH * NPIX, 256, 0, stream>>>(attn, ent);
  k_gate<<<NBH, 256, 0, stream>>>(ent, g1_w, g1_b, g2_w, g2_b, keep);

  // 7. fused top-k threshold + masked softmax (async-to-LDS staged)
  k_select_softmax<<<NBH * NPIX, 256, 0, stream>>>(attn, keep, p_h);

  // 8. P@V (2x2 register-blocked)
  k_pv<<<(NBH * 50) / 4, 128, 0, stream>>>(p_h, v_t, ctx_h);

  // 9. output projection
  k_proj<<<(NB * 16 * 25) / 4, 128, 0, stream>>>(wp_h, ctx_h, proj_b, out);
}
